// HypergraphConv_7430293422749
// MI455X (gfx1250) — matrix operand order repaired and checked by reading, compiled, not run on hardware
//
#include <hip/hip_runtime.h>
#include <hip/hip_bf16.h>

// ---------------------------------------------------------------------------
// MI455X (gfx1250) implementation of the hypergraph-conv reference.
// Heavy contractions run on v_wmma_f32_16x16x32_bf16 (fp32 accumulate).
// GEMMs: 32x32 wave tile, direct global b128 fragment loads (clean codegen:
// s_clause'd loads + wmma, no spills). Conv (211 GFLOP, dominant): 4-wave /
// 128-thread block, padded image row staged into LDS via
// GLOBAL_LOAD_ASYNC_TO_LDS_B128 (ASYNCcnt) and reused across all 7 dx taps *
// 8 K-chunks * 4 waves. Zero-padding pre-materialized in LDS so the WMMA
// path is branch-free and EXEC stays all-ones.
// ---------------------------------------------------------------------------

typedef __attribute__((ext_vector_type(16))) __bf16 v16bf;
typedef __attribute__((ext_vector_type(8)))  __bf16 v8bf;
typedef __attribute__((ext_vector_type(8)))  float  v8f;

#define BATCH 8
#define HH 64
#define WW 64
#define VV 4096          // HH*WW
#define CIN 256
#define FF 128
#define EE 256
#define COUT 256

#define XPADW 70         // 64 + 2*3 padded row width
#define LROW 264         // LDS row stride in elements (528 B): conflict-free

__device__ __forceinline__ float bf2f(__bf16 h) { return (float)h; }
__device__ __forceinline__ __bf16 f2bf(float f) { return (__bf16)f; }

__device__ __forceinline__ v8f wmma_bf16(v16bf a, v16bf b, v8f c) {
  return __builtin_amdgcn_wmma_f32_16x16x32_bf16(
      /*neg_a=*/false, a, /*neg_b=*/false, b,
      /*c_mod=*/(short)0, c, /*reuse_a=*/false, /*reuse_b=*/false);
}

__device__ __forceinline__ v16bf combine16(v8bf lo, v8bf hi) {
  v16bf r;
#pragma unroll
  for (int i = 0; i < 8; ++i) { r[i] = lo[i]; r[i + 8] = hi[i]; }
  return r;
}

// A fragment (16x32 bf16, row-major A with leading dim lda).
// lane<16: row M=lane, K {0..7},{16..23}; lane>=16: same M, K {8..15},{24..31}
__device__ __forceinline__ v16bf load_a_frag(const __bf16* A, long lda,
                                             long m0, int k0, int lane) {
  long row = m0 + (lane & 15);
  int  kb  = k0 + ((lane >> 4) << 3);     // +0 or +8
  const __bf16* p = A + row * lda + kb;
  return combine16(*(const v8bf*)(p), *(const v8bf*)(p + 16));
}

// B fragment (32x16 bf16) from B pre-transposed to [N,K] row-major (ldb = K).
// lanes 0-15: N=lane, K {0..7},{16..23} ; lanes 16-31: N=lane-16, K {8..15},{24..31}
__device__ __forceinline__ v16bf load_b_frag(const __bf16* Bt, long ldb,
                                             int n0, int k0, int lane) {
  long col = n0 + (lane & 15);
  int  kb  = k0 + ((lane >> 4) << 3);     // +0 or +8
  const __bf16* p = Bt + col * ldb + kb;
  return combine16(*(const v8bf*)(p), *(const v8bf*)(p + 16));
}

// ---------------------------------------------------------------------------
// Generic NN GEMM, B pre-transposed [N,K]; one wave computes a 32x32 tile.
// Epilogue: optional bias[n], optional (sub - acc), optional abs, optional
// f32 and/or bf16 stores. Bt may be batched along M.
// ---------------------------------------------------------------------------
__global__ void __launch_bounds__(32)
gemm_bt_kernel(const __bf16* __restrict__ A,
               const __bf16* __restrict__ Bt,
               const float*  __restrict__ bias,
               const float*  __restrict__ sub,
               float* __restrict__ outF,
               __bf16* __restrict__ outB,
               int M, int N, int K,
               int rowsPerBatch, long btBatchStride,
               int doAbs) {
  const int lane = threadIdx.x;           // blockDim.x == 32 (one wave)
  const long m0 = (long)blockIdx.x * 32;
  const int  n0 = blockIdx.y * 32;
  const __bf16* Bb = Bt + (long)(m0 / rowsPerBatch) * btBatchStride;

  v8f c00 = {}, c01 = {}, c10 = {}, c11 = {};
  for (int k0 = 0; k0 < K; k0 += 32) {
    v16bf a0 = load_a_frag(A, K, m0,      k0, lane);
    v16bf a1 = load_a_frag(A, K, m0 + 16, k0, lane);
    v16bf b0 = load_b_frag(Bb, K, n0,      k0, lane);
    v16bf b1 = load_b_frag(Bb, K, n0 + 16, k0, lane);
    c00 = wmma_bf16(a0, b0, c00);
    c01 = wmma_bf16(a0, b1, c01);
    c10 = wmma_bf16(a1, b0, c10);
    c11 = wmma_bf16(a1, b1, c11);
  }

  const int half = lane >> 4, nl = lane & 15;
  v8f acc[2][2] = { {c00, c01}, {c10, c11} };
#pragma unroll
  for (int i = 0; i < 2; ++i) {
#pragma unroll
    for (int j = 0; j < 2; ++j) {
#pragma unroll
      for (int v = 0; v < 8; ++v) {
        long m = m0 + 16 * i + 8 * half + v;
        int  n = n0 + 16 * j + nl;
        float val = acc[i][j][v];
        if (bias) val += bias[n];
        if (sub)  val = sub[m * (long)N + n] - val;
        if (doAbs) val = fabsf(val);
        long idx = m * (long)N + n;
        if (outF) outF[idx] = val;
        if (outB) outB[idx] = f2bf(val);
      }
    }
  }
}

// ---------------------------------------------------------------------------
// 7x7 SAME conv as implicit GEMM.
// Block: 128 threads (4 waves) -> 64 pixels (one full image row) x 64 E.
// Wave (wm,wn): 32 pixels x 32 E. Per dy, the interior of the padded input
// row (64 x 256 bf16) is staged into LDS with GLOBAL_LOAD_ASYNC_TO_LDS_B128
// (ASYNCcnt-tracked, no VGPR round-trip) and reused by all 7 dx taps.
// Weights (Wt: [49][E][Cin] bf16) read directly from global/L2.
// ---------------------------------------------------------------------------
__global__ void __launch_bounds__(128)
conv7_kernel(const __bf16* __restrict__ Xbf,
             const __bf16* __restrict__ Wt,
             const float*  __restrict__ bm,
             __bf16* __restrict__ Mout) {
  __shared__ __attribute__((aligned(16))) __bf16 srow[XPADW * LROW];

  const int tid  = threadIdx.x;
  const int lane = tid & 31;
  const int w    = tid >> 5;              // wave id 0..3
  const int wm   = w >> 1;                // pixel-half selector
  const int wn   = w & 1;                 // E-half selector

  const long prow = (long)blockIdx.x * 64;   // 64 pixels == one image row
  const int  b = (int)(prow >> 12);
  const int  p = (int)(prow & (VV - 1));
  const int  y = p >> 6;
  const int  e0 = blockIdx.y * 64 + wn * 32;
  const int  pxbase = wm * 32;
  const __bf16* Xb = Xbf + (long)b * VV * CIN;

  // Zero the x-padding halo once: xp in {0,1,2, 67,68,69}, 32 chunks each.
  for (int ch = tid; ch < 6 * 32; ch += 128) {
    int xpp = ch >> 5, q = ch & 31;
    int xp = (xpp < 3) ? xpp : (xpp + 64);
    v8bf z = {};
    *(v8bf*)(&srow[xp * LROW + q * 8]) = z;
  }
  __syncthreads();

  v8f c00 = {}, c01 = {}, c10 = {}, c11 = {};

  for (int dy = 0; dy < 7; ++dy) {
    const int yy = y + dy - 3;
    if (yy < 0 || yy >= HH) continue;     // block-uniform

    // Stage interior row via async global->LDS copies (16 B per lane-op):
    // 64 pixels * 32 chunks = 2048 chunks, 16 uniform iterations.
    for (int ch = tid; ch < WW * 32; ch += 128) {
      int xx = ch >> 5, q = ch & 31;
      unsigned ldsoff =
          (unsigned)(unsigned long long)(uintptr_t)(&srow[(xx + 3) * LROW + q * 8]);
      const __bf16* g = Xb + ((long)(yy * WW + xx)) * CIN + q * 8;
      asm volatile("global_load_async_to_lds_b128 %0, %1, off"
                   :: "v"(ldsoff), "v"(g)
                   : "memory");
    }
    asm volatile("s_wait_asynccnt 0x0" ::: "memory");
    __syncthreads();

    for (int dx = 0; dx < 7; ++dx) {
      const int tap = dy * 7 + dx;
      const __bf16* Wtap = Wt + (long)tap * EE * CIN;
      // Pull next tap's weight panel toward the WGP (global_prefetch_b8).
      {
        int ntap = tap < 48 ? tap + 1 : tap;
        const __bf16* pw = Wt + (long)ntap * EE * CIN + (long)(e0 + lane) * CIN;
        __builtin_prefetch((const void*)pw, 0, 0);
      }
      const int xfrag = pxbase + (lane & 15) + dx;   // padded-row index
      const int kaoff = (lane >> 4) << 3;            // +0 or +8
#pragma unroll
      for (int k0 = 0; k0 < CIN; k0 += 32) {
        const __bf16* pa0 = &srow[(xfrag)      * LROW + k0 + kaoff];
        const __bf16* pa1 = &srow[(xfrag + 16) * LROW + k0 + kaoff];
        v16bf a0 = combine16(*(const v8bf*)(pa0), *(const v8bf*)(pa0 + 16));
        v16bf a1 = combine16(*(const v8bf*)(pa1), *(const v8bf*)(pa1 + 16));
        v16bf b0 = load_b_frag(Wtap, CIN, e0,      k0, lane);
        v16bf b1 = load_b_frag(Wtap, CIN, e0 + 16, k0, lane);
        c00 = wmma_bf16(a0, b0, c00);
        c01 = wmma_bf16(a0, b1, c01);
        c10 = wmma_bf16(a1, b0, c10);
        c11 = wmma_bf16(a1, b1, c11);
      }
    }
    __syncthreads();
  }

  const int half = lane >> 4, nl = lane & 15;
  v8f acc[2][2] = { {c00, c01}, {c10, c11} };
#pragma unroll
  for (int i = 0; i < 2; ++i)
#pragma unroll
    for (int j = 0; j < 2; ++j)
#pragma unroll
      for (int v = 0; v < 8; ++v) {
        long m = prow + pxbase + 16 * i + 8 * half + v;
        int  n = e0 + 16 * j + nl;
        Mout[m * EE + n] = f2bf(acc[i][j][v] + bm[n]);
      }
}

// ---------------------------------------------------------------------------
// Prep / elementwise / reduction kernels
// ---------------------------------------------------------------------------
__global__ void cvt_bf16_kernel(const float* __restrict__ in,
                                __bf16* __restrict__ out, long n) {
  for (long i = (long)blockIdx.x * blockDim.x + threadIdx.x; i < n;
       i += (long)gridDim.x * blockDim.x)
    out[i] = f2bf(in[i]);
}

// in: [batch][K][N] f32 -> out: [batch][N][K] bf16
__global__ void trcvt_kernel(const float* __restrict__ in,
                             __bf16* __restrict__ out,
                             int K, int N, long total) {
  long kn = (long)K * N;
  for (long i = (long)blockIdx.x * blockDim.x + threadIdx.x; i < total;
       i += (long)gridDim.x * blockDim.x) {
    long t = i / kn, r = i % kn;
    int  k = (int)(r / N), n = (int)(r % N);
    out[t * kn + (long)n * K + k] = f2bf(in[i]);
  }
}

// xmean[b][c] = mean over V of x[b][v][c]; grid(B), block(CIN)
__global__ void xmean_kernel(const float* __restrict__ x,
                             float* __restrict__ xmean) {
  int b = blockIdx.x, c = threadIdx.x;
  const float* p = x + (long)b * VV * CIN + c;
  float s = 0.f;
  for (int v = 0; v < VV; ++v) s += p[(long)v * CIN];
  xmean[b * CIN + c] = s * (1.0f / VV);
}

// a[b][f] = xmean[b] @ w_a + b_a; grid(B), block(FF)
__global__ void avec_kernel(const float* __restrict__ xmean,
                            const float* __restrict__ w_a,
                            const float* __restrict__ b_a,
                            float* __restrict__ aout) {
  int b = blockIdx.x, f = threadIdx.x;
  float s = b_a[f];
  for (int c = 0; c < CIN; ++c) s += xmean[b * CIN + c] * w_a[c * FF + f];
  aout[b * FF + f] = s;
}

// t_t[b][e][f] = a[b][f] * sum_v phi[b,v,f]*M[b,v,e]; grid(E/16,F/16,B)
__global__ void t_kernel(const __bf16* __restrict__ phi_bf,
                         const __bf16* __restrict__ m_bf,
                         const float*  __restrict__ avec,
                         __bf16* __restrict__ t_t) {
  __shared__ __bf16 sp[16][17];
  __shared__ __bf16 sm[16][17];
  int b = blockIdx.z, e0 = blockIdx.x * 16, f0 = blockIdx.y * 16;
  int tx = threadIdx.x, ty = threadIdx.y;     // tx: F-local, ty: E-local
  const __bf16* P  = phi_bf + (long)b * VV * FF;
  const __bf16* Mm = m_bf   + (long)b * VV * EE;
  float acc = 0.f;
  for (int v0 = 0; v0 < VV; v0 += 16) {
    sp[ty][tx] = P[(long)(v0 + ty) * FF + f0 + tx];
    sm[ty][tx] = Mm[(long)(v0 + ty) * EE + e0 + tx];
    __syncthreads();
#pragma unroll
    for (int vv = 0; vv < 16; ++vv)
      acc += bf2f(sp[vv][tx]) * bf2f(sm[vv][ty]);
    __syncthreads();
  }
  t_t[(long)b * EE * FF + (long)(e0 + ty) * FF + f0 + tx] =
      f2bf(avec[b * FF + f0 + tx] * acc);
}

// dinv[m] = rsqrt(sum_e Hmat[m][e]); grid(B*V), block(EE)
__global__ void rowsum_rsqrt_kernel(const float* __restrict__ Hm,
                                    float* __restrict__ dinv) {
  __shared__ float s[EE];
  int m = blockIdx.x, t = threadIdx.x;
  s[t] = Hm[(long)m * EE + t];
  __syncthreads();
  for (int w = EE / 2; w > 0; w >>= 1) {
    if (t < w) s[t] += s[t + w];
    __syncthreads();
  }
  if (t == 0) dinv[m] = rsqrtf(s[0]);
}

// binv[b][e] = 1 / sum_v Hmat[b,v,e]; grid(B), block(EE)
__global__ void binv_kernel(const float* __restrict__ Hm,
                            float* __restrict__ binv) {
  int b = blockIdx.x, e = threadIdx.x;
  float s = 0.f;
  const float* p = Hm + (long)b * VV * EE + e;
  for (int v = 0; v < VV; ++v) s += p[(long)v * EE];
  binv[b * EE + e] = 1.0f / s;
}

// dh[m][e] = bf16(dinv[m] * Hmat[m][e]); grid-stride
__global__ void dh_kernel(const float* __restrict__ Hm,
                          const float* __restrict__ dinv,
                          __bf16* __restrict__ dh, long n) {
  for (long i = (long)blockIdx.x * blockDim.x + threadIdx.x; i < n;
       i += (long)gridDim.x * blockDim.x)
    dh[i] = f2bf(Hm[i] * dinv[i >> 8]);   // row = i / EE
}

// tmp_t[b][c][e] = bf16(binv[b][e] * sum_v dh[b,v,e]*x[b,v,c]); grid(E/16,C/16,B)
__global__ void tmp_kernel(const __bf16* __restrict__ dh,
                           const float*  __restrict__ x,
                           const float*  __restrict__ binv,
                           __bf16* __restrict__ tmp_t) {
  __shared__ __bf16 sd[16][17];
  __shared__ float  sx[16][17];
  int b = blockIdx.z, e0 = blockIdx.x * 16, c0 = blockIdx.y * 16;
  int tx = threadIdx.x, ty = threadIdx.y;     // tx: E-local, ty: C-local
  const __bf16* D  = dh + (long)b * VV * EE;
  const float*  X  = x  + (long)b * VV * CIN;
  float acc = 0.f;
  for (int v0 = 0; v0 < VV; v0 += 16) {
    sd[ty][tx] = D[(long)(v0 + ty) * EE + e0 + tx];
    sx[ty][tx] = X[(long)(v0 + ty) * CIN + c0 + tx];
    __syncthreads();
#pragma unroll
    for (int vv = 0; vv < 16; ++vv)
      acc += bf2f(sd[vv][tx]) * sx[vv][ty];
    __syncthreads();
  }
  tmp_t[(long)b * CIN * EE + (long)(c0 + ty) * EE + e0 + tx] =
      f2bf(binv[b * EE + e0 + tx] * acc);
}

// ---------------------------------------------------------------------------
// Host-side launch
// ---------------------------------------------------------------------------
extern "C" void kernel_launch(void* const* d_in, const int* in_sizes, int n_in,
                              void* d_out, int out_size, void* d_ws, size_t ws_size,
                              hipStream_t stream) {
  const float* x    = (const float*)d_in[0];   // [8,64,64,256]
  const float* wphi = (const float*)d_in[1];   // [256,128]
  const float* bphi = (const float*)d_in[2];   // [128]
  const float* w_a  = (const float*)d_in[3];   // [256,128]
  const float* b_a  = (const float*)d_in[4];   // [128]
  const float* w_m  = (const float*)d_in[5];   // [7,7,256,256]
  const float* b_m  = (const float*)d_in[6];   // [256]
  const float* w2   = (const float*)d_in[7];   // [256,256]
  const float* b2   = (const float*)d_in[8];   // [256]
  float* out = (float*)d_out;                  // [8,64,64,256] f32

  char* ws = (char*)d_ws;
  size_t off = 0;
  auto alloc = [&](size_t bytes) { size_t r = off; off = (off + bytes + 255) & ~(size_t)255; return r; };

  const long MROWS = (long)BATCH * VV;         // 32768
  size_t o_xbf   = alloc(MROWS * CIN * 2);     // x bf16          (aliased by r_bf later)
  size_t o_wphit = alloc((size_t)FF * CIN * 2);
  size_t o_wmt   = alloc((size_t)49 * EE * CIN * 2);
  size_t o_w2t   = alloc((size_t)COUT * CIN * 2);
  size_t o_phibf = alloc(MROWS * FF * 2);
  size_t o_avec  = alloc((size_t)BATCH * FF * 4);
  size_t o_xmean = alloc((size_t)BATCH * CIN * 4);
  size_t o_mbf   = alloc(MROWS * EE * 2);      // M bf16          (aliased by dh later)
  size_t o_tt    = alloc((size_t)BATCH * EE * FF * 2);
  size_t o_hmat  = alloc(MROWS * EE * 4);
  size_t o_dinv  = alloc(MROWS * 4);
  size_t o_binv  = alloc((size_t)BATCH * EE * 4);
  size_t o_tmpt  = alloc((size_t)BATCH * CIN * EE * 2);
  (void)ws_size; (void)in_sizes; (void)n_in; (void)out_size;

  __bf16* x_bf   = (__bf16*)(ws + o_xbf);
  __bf16* wphi_t = (__bf16*)(ws + o_wphit);
  __bf16* wm_t   = (__bf16*)(ws + o_wmt);
  __bf16* w2_t   = (__bf16*)(ws + o_w2t);
  __bf16* phi_bf = (__bf16*)(ws + o_phibf);
  float*  avec   = (float*)(ws + o_avec);
  float*  xmean  = (float*)(ws + o_xmean);
  __bf16* m_bf   = (__bf16*)(ws + o_mbf);
  __bf16* t_t    = (__bf16*)(ws + o_tt);
  float*  hmat   = (float*)(ws + o_hmat);
  float*  dinv   = (float*)(ws + o_dinv);
  float*  binv   = (float*)(ws + o_binv);
  __bf16* tmp_t  = (__bf16*)(ws + o_tmpt);
  __bf16* dh_bf  = m_bf;    // alias: M dead after t_kernel
  __bf16* r_bf   = x_bf;    // alias: x_bf dead after conv7

  // 1) Precision prep
  cvt_bf16_kernel<<<4096, 256, 0, stream>>>(x, x_bf, MROWS * CIN);
  trcvt_kernel<<<64, 256, 0, stream>>>(wphi, wphi_t, CIN, FF, (long)CIN * FF);
  trcvt_kernel<<<2048, 256, 0, stream>>>(w_m, wm_t, CIN, EE, (long)49 * CIN * EE);
  trcvt_kernel<<<128, 256, 0, stream>>>(w2, w2_t, CIN, COUT, (long)CIN * COUT);

  // 2) phi = x @ w_phi + b_phi  -> bf16 [B*V, F]
  gemm_bt_kernel<<<dim3(MROWS / 32, FF / 32), 32, 0, stream>>>(
      x_bf, wphi_t, bphi, nullptr, nullptr, phi_bf,
      (int)MROWS, FF, CIN, (int)MROWS, 0, 0);

  // 3) M = conv7x7(x, w_m) + b_m -> bf16 [B*V, E]   (dominant: ~211 GFLOP)
  conv7_kernel<<<dim3(MROWS / 64, EE / 64), 128, 0, stream>>>(x_bf, wm_t, b_m, m_bf);

  // 4) a = mean_v(x) @ w_a + b_a
  xmean_kernel<<<BATCH, CIN, 0, stream>>>(x, xmean);
  avec_kernel<<<BATCH, FF, 0, stream>>>(xmean, w_a, b_a, avec);

  // 5) t_t[b][e][f] = a[f] * (phi^T M)[f][e]
  t_kernel<<<dim3(EE / 16, FF / 16, BATCH), dim3(16, 16), 0, stream>>>(
      phi_bf, m_bf, avec, t_t);

  // 6) Hmat = |phi @ t~|  (batched B) -> f32 [B*V, E]
  gemm_bt_kernel<<<dim3(MROWS / 32, EE / 32), 32, 0, stream>>>(
      phi_bf, t_t, nullptr, nullptr, hmat, nullptr,
      (int)MROWS, EE, FF, VV, (long)EE * FF, 1);

  // 7) D^{-1/2}, B^{-1}
  rowsum_rsqrt_kernel<<<(int)MROWS, EE, 0, stream>>>(hmat, dinv);
  binv_kernel<<<BATCH, EE, 0, stream>>>(hmat, binv);

  // 8) D_H = D^{-1/2} * Hmat -> bf16 (aliases M buffer)
  dh_kernel<<<4096, 256, 0, stream>>>(hmat, dinv, dh_bf, MROWS * EE);

  // 9) tmp_t[b][c][e] = Binv[e] * (D_H^T x)[e][c]
  tmp_kernel<<<dim3(EE / 16, CIN / 16, BATCH), dim3(16, 16), 0, stream>>>(
      dh_bf, x, binv, tmp_t);

  // 10) r = x - D_H @ tmp~ -> bf16 (aliases x_bf buffer)
  gemm_bt_kernel<<<dim3(MROWS / 32, CIN / 32), 32, 0, stream>>>(
      dh_bf, tmp_t, nullptr, x, nullptr, r_bf,
      (int)MROWS, CIN, EE, VV, (long)CIN * EE, 0);

  // 11) out = r @ weight_2 + bias_2 -> f32 d_out
  gemm_bt_kernel<<<dim3(MROWS / 32, COUT / 32), 32, 0, stream>>>(
      r_bf, w2_t, b2, nullptr, out, nullptr,
      (int)MROWS, COUT, CIN, (int)MROWS, 0, 0);
}
